// SNN_DQN_6055903887853
// MI455X (gfx1250) — matrix operand order repaired and checked by reading, not compiled, not run on hardware
//
#include <hip/hip_runtime.h>

// ---------------------------------------------------------------------------
// SNN-DQN forward on gfx1250 (CDNA5, wave32, WMMA f16->f32)
// ---------------------------------------------------------------------------

#define BETA_ 0.9f
#define THR_  1.0f
#define TSTEPS 25

typedef _Float16 half_t;
typedef __attribute__((ext_vector_type(16))) _Float16 v16h;
typedef __attribute__((ext_vector_type(8)))  _Float16 v8h;
typedef __attribute__((ext_vector_type(8)))  float    v8f;

union AFrag { v16h v; v8h h[2]; };

// A fragment (M x K = 16 x 32, f16). Per ISA: lanes 0-15 hold M=lane,
// halves 0-7 -> K = kb + koff + 0..7, halves 8-15 -> K = kb + 16 + koff + 0..7,
// where koff = 8 for lanes 16-31.
__device__ __forceinline__ v16h load_a_frag(const half_t* __restrict__ S, int K,
                                            int m, int kb, int lane) {
    AFrag f;
    const int koff = (lane >> 4) << 3;            // 0 or 8
    const half_t* p = S + (size_t)m * K + kb + koff;
    f.h[0] = *(const v8h*)(p);
    f.h[1] = *(const v8h*)(p + 16);
    return f.v;
}

// B fragment (K x N = 32 x 16, f16), B[k][n] = W[n][k] (W row-major N x K).
// Lanes 0-15: N=lane, K = kb+0..7,16..23 ; lanes 16-31: K = kb+8..15,24..31 (same as A).
__device__ __forceinline__ v16h load_b_frag(const half_t* __restrict__ W, int K,
                                            int n, int kb, int lane) {
    AFrag f;
    const int koff = (lane >> 4) << 3;            // 0 or 8
    const half_t* p = W + (size_t)n * K + kb + koff;
    f.h[0] = *(const v8h*)(p);
    f.h[1] = *(const v8h*)(p + 16);
    return f.v;
}

// ---------------------------------------------------------------------------
// Fused GEMM (C = S @ W^T + bias) with optional LIF epilogue.
//   S : [M x K] f16   (spikes or x, M = 1024)
//   W : [N x K] f16
//   DO_LIF=true : mem = BETA*mem + C; spk = (mem-THR)>0; mem -= spk*THR
//   DO_LIF=false: cur_out = C
// Block: 128 threads = 4 waves (2x2). Wave tile 64x64 (4x4 WMMA 16x16):
// 16 WMMAs per K-step from only 8 fragment loads -> 2 WMMA per fragment.
// Block tile 128(M) x 128(N). Grid: (N/128, M/128).
// ---------------------------------------------------------------------------
template <bool DO_LIF>
__global__ __launch_bounds__(128)
void gemm_lif_kernel(const half_t* __restrict__ S, const half_t* __restrict__ W,
                     const float* __restrict__ bias, float* __restrict__ mem,
                     half_t* __restrict__ spk_out, float* __restrict__ cur_out,
                     int K, int N)
{
    const int lane = threadIdx.x & 31;
    const int wave = threadIdx.x >> 5;
    const int wm = wave & 1;               // 0..1 -> M direction
    const int wn = wave >> 1;              // 0..1 -> N direction
    const int m0 = blockIdx.y * 128 + wm * 64;
    const int n0 = blockIdx.x * 128 + wn * 64;
    const int l15 = lane & 15;

    v8f zero = {};
    v8f acc[4][4];
#pragma unroll
    for (int i = 0; i < 4; ++i)
#pragma unroll
        for (int j = 0; j < 4; ++j) acc[i][j] = zero;

    for (int kb = 0; kb < K; kb += 32) {
        v16h a[4], b[4];
#pragma unroll
        for (int i = 0; i < 4; ++i)
            a[i] = load_a_frag(S, K, m0 + i * 16 + l15, kb, lane);
#pragma unroll
        for (int j = 0; j < 4; ++j)
            b[j] = load_b_frag(W, K, n0 + j * 16 + l15, kb, lane);
#pragma unroll
        for (int i = 0; i < 4; ++i)
#pragma unroll
            for (int j = 0; j < 4; ++j)
                acc[i][j] = __builtin_amdgcn_wmma_f32_16x16x32_f16(
                    false, a[i], false, b[j], (short)0, acc[i][j], false, false);
    }

    // C layout: component r at lane L -> row = r + 8*(L>=16), col = L&15
    const int rowAdd = (lane >> 4) << 3;
#pragma unroll
    for (int i = 0; i < 4; ++i) {
#pragma unroll
        for (int j = 0; j < 4; ++j) {
            const int col = n0 + j * 16 + l15;
            const float bb = bias[col];
#pragma unroll
            for (int r = 0; r < 8; ++r) {
                const int row = m0 + i * 16 + rowAdd + r;
                const size_t idx = (size_t)row * N + col;
                const float c = acc[i][j][r] + bb;
                if (DO_LIF) {
                    float mval = BETA_ * mem[idx] + c;
                    float sp = (mval - THR_) > 0.0f ? 1.0f : 0.0f;
                    mem[idx] = mval - sp * THR_;
                    spk_out[idx] = (half_t)sp;
                } else {
                    cur_out[idx] = c;
                }
            }
        }
    }
}

// ---------------------------------------------------------------------------
// Dueling head: Whead [32 x K] f16 (rows 0..17 = Wa, row 18 = Wv, rest 0).
// Per block: 128 rows, 4 waves each doing 32x32 WMMA; result staged in LDS;
// each thread finalizes one batch row: q = v + a - max(a); accq += q.
// Grid: (B/128). Block 128 threads.
// ---------------------------------------------------------------------------
__global__ __launch_bounds__(128)
void head_kernel(const half_t* __restrict__ S, const half_t* __restrict__ Wh,
                 const float* __restrict__ ba, const float* __restrict__ bvv,
                 float* __restrict__ accq, int K)
{
    __shared__ float lds[128 * 34];

    const int lane = threadIdx.x & 31;
    const int wave = threadIdx.x >> 5;
    const int blockM = blockIdx.x * 128;
    const int m0 = blockM + wave * 32;
    const int l15 = lane & 15;

    v8f zero = {};
    v8f acc[2][2];
#pragma unroll
    for (int i = 0; i < 2; ++i)
#pragma unroll
        for (int j = 0; j < 2; ++j) acc[i][j] = zero;

    for (int kb = 0; kb < K; kb += 32) {
        v16h a[2], b[2];
#pragma unroll
        for (int i = 0; i < 2; ++i)
            a[i] = load_a_frag(S, K, m0 + i * 16 + l15, kb, lane);
#pragma unroll
        for (int j = 0; j < 2; ++j)
            b[j] = load_b_frag(Wh, K, j * 16 + l15, kb, lane);
#pragma unroll
        for (int i = 0; i < 2; ++i)
#pragma unroll
            for (int j = 0; j < 2; ++j)
                acc[i][j] = __builtin_amdgcn_wmma_f32_16x16x32_f16(
                    false, a[i], false, b[j], (short)0, acc[i][j], false, false);
    }

    const int rowAdd = (lane >> 4) << 3;
#pragma unroll
    for (int i = 0; i < 2; ++i)
#pragma unroll
        for (int j = 0; j < 2; ++j)
#pragma unroll
            for (int r = 0; r < 8; ++r) {
                const int rl = wave * 32 + i * 16 + rowAdd + r;
                const int cl = j * 16 + l15;
                lds[rl * 34 + cl] = acc[i][j][r];
            }
    __syncthreads();

    // one thread per batch row
    const int rl = threadIdx.x;
    const int row = blockM + rl;
    const float vhead = lds[rl * 34 + 18] + bvv[0];
    float av[18];
    float amax = -3.0e38f;
#pragma unroll
    for (int n = 0; n < 18; ++n) {
        av[n] = lds[rl * 34 + n] + ba[n];
        amax = fmaxf(amax, av[n]);
    }
#pragma unroll
    for (int n = 0; n < 18; ++n)
        accq[(size_t)row * 18 + n] += vhead + av[n] - amax;
}

// ---------------------------------------------------------------------------
// Small utility kernels
// ---------------------------------------------------------------------------
__global__ void lif1_kernel(const float* __restrict__ cur, float* __restrict__ mem,
                            half_t* __restrict__ spk, int n)
{
    int i = blockIdx.x * blockDim.x + threadIdx.x;
    if (i < n) {
        float mval = BETA_ * mem[i] + cur[i];
        float sp = (mval - THR_) > 0.0f ? 1.0f : 0.0f;
        mem[i] = mval - sp * THR_;
        spk[i] = (half_t)sp;
    }
}

__global__ void f32_to_f16_kernel(const float* __restrict__ in,
                                  half_t* __restrict__ out, int n)
{
    int i = blockIdx.x * blockDim.x + threadIdx.x;
    if (i < n) out[i] = (half_t)in[i];
}

__global__ void pack_head_kernel(const float* __restrict__ Wa,
                                 const float* __restrict__ Wv,
                                 half_t* __restrict__ out, int K)
{
    int i = blockIdx.x * blockDim.x + threadIdx.x;
    if (i < 32 * K) {
        int r = i / K, k = i - r * K;
        float v = (r < 18) ? Wa[(size_t)r * K + k] : (r == 18 ? Wv[k] : 0.0f);
        out[i] = (half_t)v;
    }
}

__global__ void zero_f32_kernel(float* __restrict__ p, int n)
{
    int i = blockIdx.x * blockDim.x + threadIdx.x;
    if (i < n) p[i] = 0.0f;
}

__global__ void finalize_kernel(const float* __restrict__ acc,
                                float* __restrict__ out, int n)
{
    int i = blockIdx.x * blockDim.x + threadIdx.x;
    if (i < n) out[i] = acc[i] * (1.0f / (float)TSTEPS);
}

// ---------------------------------------------------------------------------
// Launcher
// ---------------------------------------------------------------------------
extern "C" void kernel_launch(void* const* d_in, const int* in_sizes, int n_in,
                              void* d_out, int out_size, void* d_ws, size_t ws_size,
                              hipStream_t stream)
{
    (void)in_sizes; (void)n_in; (void)out_size; (void)ws_size;
    const int B = 1024, IN = 1024, H = 2048, A = 18;

    const float* x  = (const float*)d_in[0];
    const float* W1 = (const float*)d_in[1];
    const float* b1 = (const float*)d_in[2];
    const float* W2 = (const float*)d_in[3];
    const float* b2 = (const float*)d_in[4];
    const float* W3 = (const float*)d_in[5];
    const float* b3 = (const float*)d_in[6];
    const float* W4 = (const float*)d_in[7];
    const float* b4 = (const float*)d_in[8];
    const float* W5 = (const float*)d_in[9];
    const float* b5 = (const float*)d_in[10];
    const float* Wv = (const float*)d_in[11];
    const float* bv = (const float*)d_in[12];
    const float* Wa = (const float*)d_in[13];
    const float* ba = (const float*)d_in[14];

    char* ws = (char*)d_ws;
    size_t off = 0;
    auto carve = [&](size_t bytes) -> void* {
        void* p = ws + off;
        off += (bytes + 255) & ~(size_t)255;
        return p;
    };

    half_t* W1h = (half_t*)carve((size_t)H * IN * 2);
    half_t* W2h = (half_t*)carve((size_t)H * H * 2);
    half_t* W3h = (half_t*)carve((size_t)H * H * 2);
    half_t* W4h = (half_t*)carve((size_t)H * H * 2);
    half_t* W5h = (half_t*)carve((size_t)H * H * 2);
    half_t* Whd = (half_t*)carve((size_t)32 * H * 2);
    half_t* xh  = (half_t*)carve((size_t)B * IN * 2);
    float*  cur1 = (float*)carve((size_t)B * H * 4);
    float*  mem1 = (float*)carve((size_t)B * H * 4);
    float*  mem2 = (float*)carve((size_t)B * H * 4);
    float*  mem3 = (float*)carve((size_t)B * H * 4);
    float*  mem4 = (float*)carve((size_t)B * H * 4);
    float*  mem5 = (float*)carve((size_t)B * H * 4);
    half_t* sA  = (half_t*)carve((size_t)B * H * 2);
    half_t* sB  = (half_t*)carve((size_t)B * H * 2);
    float*  acc = (float*)carve((size_t)B * A * 4);

    const int TB = 256;
    auto blocks = [&](int n) { return (n + TB - 1) / TB; };

    // --- prep: f16 conversions, head packing, state init ---
    f32_to_f16_kernel<<<blocks(H * IN), TB, 0, stream>>>(W1, W1h, H * IN);
    f32_to_f16_kernel<<<blocks(H * H), TB, 0, stream>>>(W2, W2h, H * H);
    f32_to_f16_kernel<<<blocks(H * H), TB, 0, stream>>>(W3, W3h, H * H);
    f32_to_f16_kernel<<<blocks(H * H), TB, 0, stream>>>(W4, W4h, H * H);
    f32_to_f16_kernel<<<blocks(H * H), TB, 0, stream>>>(W5, W5h, H * H);
    f32_to_f16_kernel<<<blocks(B * IN), TB, 0, stream>>>(x, xh, B * IN);
    pack_head_kernel<<<blocks(32 * H), TB, 0, stream>>>(Wa, Wv, Whd, H);
    zero_f32_kernel<<<blocks(B * H), TB, 0, stream>>>(mem1, B * H);
    zero_f32_kernel<<<blocks(B * H), TB, 0, stream>>>(mem2, B * H);
    zero_f32_kernel<<<blocks(B * H), TB, 0, stream>>>(mem3, B * H);
    zero_f32_kernel<<<blocks(B * H), TB, 0, stream>>>(mem4, B * H);
    zero_f32_kernel<<<blocks(B * H), TB, 0, stream>>>(mem5, B * H);
    zero_f32_kernel<<<blocks(B * A), TB, 0, stream>>>(acc, B * A);

    // --- layer-1 current is time-invariant: compute once ---
    dim3 g1(H / 128, B / 128);
    gemm_lif_kernel<false><<<g1, 128, 0, stream>>>(xh, W1h, b1, nullptr, nullptr,
                                                   cur1, IN, H);

    // --- timestep loop ---
    dim3 gh(H / 128, B / 128);
    for (int t = 0; t < TSTEPS; ++t) {
        lif1_kernel<<<blocks(B * H), TB, 0, stream>>>(cur1, mem1, sA, B * H);
        gemm_lif_kernel<true><<<gh, 128, 0, stream>>>(sA, W2h, b2, mem2, sB,
                                                      nullptr, H, H);
        gemm_lif_kernel<true><<<gh, 128, 0, stream>>>(sB, W3h, b3, mem3, sA,
                                                      nullptr, H, H);
        gemm_lif_kernel<true><<<gh, 128, 0, stream>>>(sA, W4h, b4, mem4, sB,
                                                      nullptr, H, H);
        gemm_lif_kernel<true><<<gh, 128, 0, stream>>>(sB, W5h, b5, mem5, sA,
                                                      nullptr, H, H);
        head_kernel<<<B / 128, 128, 0, stream>>>(sA, Whd, ba, bv, acc, H);
    }

    finalize_kernel<<<blocks(B * A), TB, 0, stream>>>(acc, (float*)d_out, B * A);
}
